// AttentionHead_26448408609009
// MI455X (gfx1250) — matrix operand order repaired and checked by reading, not compiled, not run on hardware
//
#include <hip/hip_runtime.h>
#include <hip/hip_bf16.h>
#include <math.h>

typedef __bf16 bf16_t;
typedef __attribute__((ext_vector_type(8)))  bf16_t bf16x8;
typedef __attribute__((ext_vector_type(16))) bf16_t bf16x16;
typedef __attribute__((ext_vector_type(8)))  float  f32x8;

#define B_DIM  64
#define N_SEQ  1024
#define DH     512
#define E_DIM  1024   // 2*DH

// CDNA5 async global->LDS staging (ASYNCcnt) if the toolchain exposes it.
#if defined(__AMDGCN__) && __has_builtin(__builtin_amdgcn_global_load_async_to_lds_b128) && \
    __has_builtin(__builtin_amdgcn_s_wait_asynccnt)
#define USE_ASYNC_LDS 1
#else
#define USE_ASYNC_LDS 0
#endif

#if USE_ASYNC_LDS
typedef int int4v __attribute__((vector_size(16)));
typedef __attribute__((address_space(1))) int4v* g_b128_ptr;  // global src
typedef __attribute__((address_space(3))) int4v* l_b128_ptr;  // LDS dst

__device__ __forceinline__ void async_cp16(const bf16_t* g, bf16_t* l) {
  __builtin_amdgcn_global_load_async_to_lds_b128(
      (g_b128_ptr)g, (l_b128_ptr)l, 0, 0);
}
#endif

// ---------------------------------------------------------------------------
// f32 -> bf16 conversion
// ---------------------------------------------------------------------------
__global__ __launch_bounds__(256) void cvt_f32_to_bf16(const float* __restrict__ in,
                                                       bf16_t* __restrict__ out, long n) {
  long i = (long)blockIdx.x * blockDim.x + threadIdx.x;
  if (i < n) out[i] = (bf16_t)in[i];
}

// f32 [R][C] -> bf16 transposed [C][R]
__global__ __launch_bounds__(256) void cvt_transpose_f32_bf16(const float* __restrict__ in,
                                                              bf16_t* __restrict__ out,
                                                              int R, int C) {
  long i = (long)blockIdx.x * blockDim.x + threadIdx.x;
  if (i < (long)R * C) {
    int r = (int)(i / C), c = (int)(i % C);
    out[(long)c * R + r] = (bf16_t)in[i];
  }
}

// ---------------------------------------------------------------------------
// bf16 WMMA GEMM, C[M,N] = A[M,K] * B, where B is stored "K-contiguous":
// element (k,n) at Bptr[n*ldb + k].  Block tile 128x128, K-step 32, 8 wave32s
// (4x2), each wave 2x4 tiles of v_wmma_f32_16x16x32_bf16.  Double-buffered LDS
// with async global->LDS copies when available.
// MODE 0: +bias; store bf16 C (qv), bf16 relu(C) (kb), and bf16 C transposed
//         per batch (vT)                                   [fc1]
// MODE 1: store f32 C*scale                                [attention logits]
// MODE 2: store bf16 C                                     [P @ V]
// MODE 3: store f32 relu(C + bias)                         [fc2 -> output]
// ---------------------------------------------------------------------------
constexpr int BM = 128, BN = 128, BK = 32, PAD = 8;
constexpr int LDK = BK + PAD;  // 40 halves = 80B row stride (16B aligned)

template <int MODE>
__global__ __launch_bounds__(256) void gemm_bf16_wmma(
    const bf16_t* __restrict__ A, const bf16_t* __restrict__ B,
    const float* __restrict__ bias, float scale,
    float* __restrict__ outF, bf16_t* __restrict__ outB0, bf16_t* __restrict__ outB1,
    bf16_t* __restrict__ outBT,
    int K, int lda, int ldb, int ldo,
    long batA, long batB, long batO) {
  __shared__ bf16_t Ash[2][BM][LDK];
  __shared__ bf16_t Bsh[2][BN][LDK];

  const int  m0 = blockIdx.y * BM;
  const int  n0 = blockIdx.x * BN;
  const long zb = blockIdx.z;
  A += zb * batA;
  B += zb * batB;

  const int tid  = threadIdx.x;
  const int lane = tid & 31;
  const int wave = tid >> 5;
  const int wm   = wave >> 1;  // 0..3 -> 32 rows
  const int wn   = wave & 1;   // 0..1 -> 64 cols
  const int g    = lane >> 4;
  const int ls   = lane & 15;

  // staging: thread owns rows sr and sr+64 of both tiles, 8-half chunk sc
  const int sr = tid >> 2;        // 0..63
  const int sc = (tid & 3) * 8;   // 0,8,16,24
  const bf16_t* gA0 = A + (long)(m0 + sr) * lda + sc;
  const bf16_t* gA1 = A + (long)(m0 + sr + 64) * lda + sc;
  const bf16_t* gB0 = B + (long)(n0 + sr) * ldb + sc;
  const bf16_t* gB1 = B + (long)(n0 + sr + 64) * ldb + sc;

  f32x8 acc[2][4];
#pragma unroll
  for (int i = 0; i < 2; ++i)
#pragma unroll
    for (int j = 0; j < 4; ++j)
#pragma unroll
      for (int r = 0; r < 8; ++r) acc[i][j][r] = 0.0f;

  auto compute = [&](int buf) {
    bf16x16 af[2], bfrag[4];
#pragma unroll
    for (int i = 0; i < 2; ++i) {
      int r = wm * 32 + i * 16 + ls;
      bf16x8 lo = *(const bf16x8*)&Ash[buf][r][8 * g];       // K = 8g..8g+7
      bf16x8 hi = *(const bf16x8*)&Ash[buf][r][16 + 8 * g];  // K = 16+8g..23+8g
      af[i] = __builtin_shufflevector(lo, hi, 0, 1, 2, 3, 4, 5, 6, 7,
                                      8, 9, 10, 11, 12, 13, 14, 15);
    }
#pragma unroll
    for (int j = 0; j < 4; ++j) {
      int c = wn * 64 + j * 16 + ls;
      bf16x8 lo = *(const bf16x8*)&Bsh[buf][c][8 * g];       // K = 8g..8g+7
      bf16x8 hi = *(const bf16x8*)&Bsh[buf][c][16 + 8 * g];  // K = 16+8g..23+8g
      bfrag[j] = __builtin_shufflevector(lo, hi, 0, 1, 2, 3, 4, 5, 6, 7,
                                         8, 9, 10, 11, 12, 13, 14, 15);
    }
#pragma unroll
    for (int i = 0; i < 2; ++i)
#pragma unroll
      for (int j = 0; j < 4; ++j)
        acc[i][j] = __builtin_amdgcn_wmma_f32_16x16x32_bf16(
            false, af[i], false, bfrag[j], (short)0, acc[i][j], false, false);
  };

  const int nk = K / BK;

#if USE_ASYNC_LDS
  // prologue: async-stage tile 0 into buffer 0
  async_cp16(gA0, &Ash[0][sr][sc]);
  async_cp16(gA1, &Ash[0][sr + 64][sc]);
  async_cp16(gB0, &Bsh[0][sr][sc]);
  async_cp16(gB1, &Bsh[0][sr + 64][sc]);
  gA0 += BK; gA1 += BK; gB0 += BK; gB1 += BK;
  __builtin_amdgcn_s_wait_asynccnt(0);
  __syncthreads();
  for (int kt = 0; kt < nk; ++kt) {
    const int  cur     = kt & 1;
    const bool hasNext = (kt + 1) < nk;
    if (hasNext) {  // overlap next-tile DMA with this tile's WMMAs
      const int nxt = 1 - cur;
      async_cp16(gA0, &Ash[nxt][sr][sc]);
      async_cp16(gA1, &Ash[nxt][sr + 64][sc]);
      async_cp16(gB0, &Bsh[nxt][sr][sc]);
      async_cp16(gB1, &Bsh[nxt][sr + 64][sc]);
      gA0 += BK; gA1 += BK; gB0 += BK; gB1 += BK;
    }
    compute(cur);
    if (hasNext) __builtin_amdgcn_s_wait_asynccnt(0);
    __syncthreads();
  }
#else
  for (int kt = 0; kt < nk; ++kt) {
    bf16x8 ra0 = *(const bf16x8*)gA0;
    bf16x8 ra1 = *(const bf16x8*)gA1;
    bf16x8 rb0 = *(const bf16x8*)gB0;
    bf16x8 rb1 = *(const bf16x8*)gB1;
    gA0 += BK; gA1 += BK; gB0 += BK; gB1 += BK;
    __syncthreads();  // prior reads of buffer 0 done
    *(bf16x8*)&Ash[0][sr][sc]      = ra0;
    *(bf16x8*)&Ash[0][sr + 64][sc] = ra1;
    *(bf16x8*)&Bsh[0][sr][sc]      = rb0;
    *(bf16x8*)&Bsh[0][sr + 64][sc] = rb1;
    __syncthreads();
    compute(0);
  }
#endif

  // epilogue; C/D layout: row = base + r + 8*(lane>>4), col = base + (lane&15)
#pragma unroll
  for (int i = 0; i < 2; ++i) {
#pragma unroll
    for (int j = 0; j < 4; ++j) {
      const int rowb = m0 + wm * 32 + i * 16 + g * 8;
      const int col  = n0 + wn * 64 + j * 16 + ls;
      if constexpr (MODE == 0) {
        bf16x8 tv;
#pragma unroll
        for (int r = 0; r < 8; ++r) {
          float val = acc[i][j][r] + bias[col];
          long  off = (long)(rowb + r) * ldo + col;
          outB0[off] = (bf16_t)val;
          outB1[off] = (bf16_t)fmaxf(val, 0.0f);
          tv[r] = (bf16_t)val;
        }
        // V^T store: [batch][col][row%1024], rows contiguous -> one 16B store
        const long b  = rowb >> 10;
        const long nn = rowb & (N_SEQ - 1);
        *(bf16x8*)(outBT + b * (long)E_DIM * N_SEQ + (long)col * N_SEQ + nn) = tv;
      } else {
#pragma unroll
        for (int r = 0; r < 8; ++r) {
          long  off = (long)(rowb + r) * ldo + col + zb * batO;
          float val = acc[i][j][r];
          if constexpr (MODE == 1) {
            outF[off] = val * scale;
          } else if constexpr (MODE == 2) {
            outB0[off] = (bf16_t)val;
          } else {
            outF[off] = fmaxf(val + bias[col], 0.0f);
          }
        }
      }
    }
  }
}

// ---------------------------------------------------------------------------
// Row softmax over S[.,1024] -> bf16 P.  One 256-thread block per row.
// ---------------------------------------------------------------------------
__global__ __launch_bounds__(256) void softmax_rows(const float* __restrict__ S,
                                                    bf16_t* __restrict__ P) {
  const long row = blockIdx.x;
  const float* s = S + row * (long)N_SEQ;
  bf16_t* p = P + row * (long)N_SEQ;
  const int tid = threadIdx.x, lane = tid & 31, wave = tid >> 5;
  __shared__ float red[8];

  float v[4];
#pragma unroll
  for (int i = 0; i < 4; ++i) v[i] = s[tid + i * 256];

  float m = fmaxf(fmaxf(v[0], v[1]), fmaxf(v[2], v[3]));
#pragma unroll
  for (int off = 16; off; off >>= 1) m = fmaxf(m, __shfl_xor(m, off, 32));
  if (lane == 0) red[wave] = m;
  __syncthreads();
  float rowmax = red[0];
#pragma unroll
  for (int i = 1; i < 8; ++i) rowmax = fmaxf(rowmax, red[i]);
  __syncthreads();

  float e[4], sum = 0.0f;
#pragma unroll
  for (int i = 0; i < 4; ++i) { e[i] = __expf(v[i] - rowmax); sum += e[i]; }
#pragma unroll
  for (int off = 16; off; off >>= 1) sum += __shfl_xor(sum, off, 32);
  if (lane == 0) red[wave] = sum;
  __syncthreads();
  float tot = 0.0f;
#pragma unroll
  for (int i = 0; i < 8; ++i) tot += red[i];
  float inv = 1.0f / tot;
#pragma unroll
  for (int i = 0; i < 4; ++i) p[tid + i * 256] = (bf16_t)(e[i] * inv);
}

// ---------------------------------------------------------------------------
extern "C" void kernel_launch(void* const* d_in, const int* in_sizes, int n_in,
                              void* d_out, int out_size, void* d_ws, size_t ws_size,
                              hipStream_t stream) {
  const float* x     = (const float*)d_in[0];
  const float* W1    = (const float*)d_in[1];
  const float* bias1 = (const float*)d_in[2];
  const float* W2    = (const float*)d_in[3];
  const float* bias2 = (const float*)d_in[4];
  float* out = (float*)d_out;

  const long MROWS = (long)B_DIM * N_SEQ;  // 65536
  const long XE    = MROWS * DH;
  const long PE    = MROWS * E_DIM;
  const long SE    = (long)B_DIM * N_SEQ * N_SEQ;

  char* w = (char*)d_ws;
  auto take = [&](size_t bytes) -> char* {
    char* p = w;
    w += (bytes + 255) & ~(size_t)255;
    return p;
  };
  bf16_t* xb   = (bf16_t*)take((size_t)XE * 2);
  bf16_t* w1t  = (bf16_t*)take((size_t)DH * E_DIM * 2);   // [E][DH]
  bf16_t* w2t  = (bf16_t*)take((size_t)E_DIM * DH * 2);   // [DH][E]
  bf16_t* qv   = (bf16_t*)take((size_t)PE * 2);           // proj (Q, row-major)
  bf16_t* kb   = (bf16_t*)take((size_t)PE * 2);           // relu(proj) (K)
  bf16_t* vT   = (bf16_t*)take((size_t)PE * 2);           // proj^T per batch (V^T)
  float*  Smat = (float*)take((size_t)SE * 4);
  bf16_t* Pmat = (bf16_t*)take((size_t)SE * 2);
  bf16_t* feat = (bf16_t*)take((size_t)PE * 2);

  cvt_f32_to_bf16<<<dim3((unsigned)((XE + 255) / 256)), 256, 0, stream>>>(x, xb, XE);
  cvt_transpose_f32_bf16<<<dim3((DH * E_DIM + 255) / 256), 256, 0, stream>>>(W1, w1t, DH, E_DIM);
  cvt_transpose_f32_bf16<<<dim3((E_DIM * DH + 255) / 256), 256, 0, stream>>>(W2, w2t, E_DIM, DH);

  // fc1: proj = x@W1 + b1 -> qv, kb = relu, vT = proj^T per batch
  gemm_bf16_wmma<0><<<dim3(E_DIM / BN, MROWS / BM, 1), 256, 0, stream>>>(
      xb, w1t, bias1, 1.0f, nullptr, qv, kb, vT,
      DH, DH, DH, E_DIM, 0, 0, 0);

  // S = (Q @ K^T) / sqrt(2d), batched over 64
  gemm_bf16_wmma<1><<<dim3(N_SEQ / BN, N_SEQ / BM, B_DIM), 256, 0, stream>>>(
      qv, kb, nullptr, 1.0f / 32.0f, Smat, nullptr, nullptr, nullptr,
      E_DIM, E_DIM, E_DIM, N_SEQ,
      (long)N_SEQ * E_DIM, (long)N_SEQ * E_DIM, (long)N_SEQ * N_SEQ);

  softmax_rows<<<dim3((unsigned)(B_DIM * N_SEQ)), 256, 0, stream>>>(Smat, Pmat);

  // feature = P @ V  (B = V^T, K-contiguous), batched
  gemm_bf16_wmma<2><<<dim3(E_DIM / BN, N_SEQ / BM, B_DIM), 256, 0, stream>>>(
      Pmat, vT, nullptr, 1.0f, nullptr, feat, nullptr, nullptr,
      N_SEQ, N_SEQ, N_SEQ, E_DIM,
      (long)N_SEQ * N_SEQ, (long)E_DIM * N_SEQ, (long)N_SEQ * E_DIM);

  // out = relu(feature @ W2 + b2), f32
  gemm_bf16_wmma<3><<<dim3(DH / BN, MROWS / BM, 1), 256, 0, stream>>>(
      feat, w2t, bias2, 1.0f, out, nullptr, nullptr, nullptr,
      E_DIM, E_DIM, E_DIM, DH, 0, 0, 0);
}
